// DeepSetQNetwork_24094766531466
// MI455X (gfx1250) — matrix operand order repaired and checked by reading, not compiled, not run on hardware
//
#include <hip/hip_runtime.h>

#define B_TOTAL 524288
#define NSET 12
#define DDYN 4
#define DSTA 7

typedef __attribute__((ext_vector_type(16))) __bf16 bf16x16;
typedef __attribute__((ext_vector_type(8)))  float  f32x8;
typedef __attribute__((ext_vector_type(4)))  float  f32x4;

// hardware bf16 convert (lowers to v_cvt_pk_bf16_f32 / v_cvt_bf16_f32, RNE)
static __device__ __forceinline__ unsigned short f2bf(float f) {
  return __builtin_bit_cast(unsigned short, static_cast<__bf16>(f));
}
static __device__ __forceinline__ __bf16 bfb(unsigned short s) {
  return __builtin_bit_cast(__bf16, s);
}

// A fragment (16xK bf16): row = lane%16; in-frag k' = j + 8*(lane>=16) + 8*(j>=8).
// Source is row-major [16][ld] bf16 in LDS; absolute K range [kbase, kbase+32), zero pad >= kvalid.
static __device__ __forceinline__ bf16x16 load_a(const unsigned short* act, int ld,
                                                 int kbase, int kvalid) {
  const int lane = threadIdx.x & 31;
  const int row  = lane & 15;
  const int khi  = (lane >> 4) << 3;
  bf16x16 a;
#pragma unroll
  for (int j = 0; j < 16; ++j) {
    int k = kbase + j + khi + (j & 8);
    unsigned short v = (k < kvalid) ? act[row * ld + k] : (unsigned short)0;
    a[j] = bfb(v);
  }
  return a;
}

// B fragment (Kx16 bf16): col = n0 + lane%16; K = kbase + j + 8*(lane>=16) + 8*(j>=8).
// Source row-major [krows][n] bf16 in LDS; zero pad K >= krows.
static __device__ __forceinline__ bf16x16 load_b(const unsigned short* w, int n, int krows,
                                                 int kbase, int n0) {
  const int lane = threadIdx.x & 31;
  const int col  = n0 + (lane & 15);
  const int kb   = kbase + ((lane >> 4) << 3);
  bf16x16 b;
#pragma unroll
  for (int j = 0; j < 16; ++j) {
    int k = kb + j + (j & 8);
    unsigned short v = (k < krows) ? w[k * n + col] : (unsigned short)0;
    b[j] = bfb(v);
  }
  return b;
}

static __device__ __forceinline__ f32x8 wmma_bf16(bf16x16 a, bf16x16 b, f32x8 c) {
  return __builtin_amdgcn_wmma_f32_16x16x32_bf16(false, a, false, b, (short)0, c, false, false);
}

// C/D fragment: col = n0 + lane%16, rows = 8*(lane>=16) + r.  bias + optional relu -> bf16 LDS.
static __device__ __forceinline__ void store_c(unsigned short* dst, int ld, int n0,
                                               f32x8 c, const float* bias, bool relu) {
  const int lane = threadIdx.x & 31;
  const int col  = n0 + (lane & 15);
  const int rb   = (lane >> 4) << 3;
  float bb = bias ? bias[col] : 0.0f;
#pragma unroll
  for (int r = 0; r < 8; ++r) {
    float v = c[r] + bb;
    if (relu) v = fmaxf(v, 0.0f);
    dst[(rb + r) * ld + col] = f2bf(v);
  }
}

static __device__ __forceinline__ void relu_bias_acc(f32x8& acc, f32x8 c,
                                                     const float* bias, int n0) {
  const int lane = threadIdx.x & 31;
  float bb = bias[n0 + (lane & 15)];
#pragma unroll
  for (int r = 0; r < 8; ++r) acc[r] += fmaxf(c[r] + bb, 0.0f);
}

__global__ __launch_bounds__(256)
void deepset_q_kernel(const float* __restrict__ dyn, const float* __restrict__ sta,
                      const float* __restrict__ w1,  const float* __restrict__ b1,
                      const float* __restrict__ w2,  const float* __restrict__ b2,
                      const float* __restrict__ wr1, const float* __restrict__ br1,
                      const float* __restrict__ wr2, const float* __restrict__ br2,
                      const float* __restrict__ wq1, const float* __restrict__ bq1,
                      const float* __restrict__ wq2, const float* __restrict__ bq2,
                      const float* __restrict__ wo,  const float* __restrict__ bo,
                      float* __restrict__ out) {
  __shared__ unsigned short s_w1[4 * 16];
  __shared__ unsigned short s_w2[16 * 32];
  __shared__ unsigned short s_wr1[32 * 32];
  __shared__ unsigned short s_wr2[32 * 16];
  __shared__ unsigned short s_wq1[32 * 64];   // rows 23..31 zero-padded
  __shared__ unsigned short s_wq2[64 * 64];
  __shared__ unsigned short s_wo[64 * 16];    // cols 5..15 zero-padded
  __shared__ float s_b1[16], s_b2[32], s_br1[32], s_br2[16], s_bq1[64], s_bq2[64], s_bo[16];
  __shared__ unsigned short bounce[8][16 * 64];  // per-wave activation tile

  const int tid = threadIdx.x;

  // ---- stage weights (fp32 -> bf16) and biases into LDS, once per block ----
  for (int i = tid; i < 64;   i += 256) s_w1[i]  = f2bf(w1[i]);
  for (int i = tid; i < 512;  i += 256) s_w2[i]  = f2bf(w2[i]);
  for (int i = tid; i < 1024; i += 256) s_wr1[i] = f2bf(wr1[i]);
  for (int i = tid; i < 512;  i += 256) s_wr2[i] = f2bf(wr2[i]);
  for (int i = tid; i < 2048; i += 256) {
    int r = i >> 6, c = i & 63;
    s_wq1[i] = (r < 23) ? f2bf(wq1[r * 64 + c]) : (unsigned short)0;
  }
  for (int i = tid; i < 4096; i += 256) s_wq2[i] = f2bf(wq2[i]);
  for (int i = tid; i < 1024; i += 256) {
    int r = i >> 4, c = i & 15;
    s_wo[i] = (c < 5) ? f2bf(wo[r * 5 + c]) : (unsigned short)0;
  }
  if (tid < 16) s_b1[tid]  = b1[tid];
  if (tid < 32) s_b2[tid]  = b2[tid];
  if (tid < 32) s_br1[tid] = br1[tid];
  if (tid < 16) s_br2[tid] = br2[tid];
  if (tid < 64) s_bq1[tid] = bq1[tid];
  if (tid < 64) s_bq2[tid] = bq2[tid];
  if (tid < 16) s_bo[tid]  = (tid < 5) ? bo[tid] : 0.0f;
  __syncthreads();

  const int wave = tid >> 5;
  const int lane = tid & 31;
  const int row  = lane & 15;
  const long s0  = ((long)blockIdx.x * 8 + wave) * 16;   // 16 samples per wave
  unsigned short* myb = bounce[wave];

  // gfx1250 prefetch of the static rows we need later (global_prefetch_b8)
  __builtin_prefetch(&sta[(s0 + row) * DSTA], 0, 3);

  const f32x8 zc = {0, 0, 0, 0, 0, 0, 0, 0};

  // ---- phi stage, fused phi1->phi2->relu->set-sum ----
  f32x8 acc0 = zc, acc1 = zc;                 // x_sum cols 0..15 / 16..31
  bf16x16 bphi1  = load_b(s_w1, 16, 4, 0, 0);
  bf16x16 bphi2a = load_b(s_w2, 32, 16, 0, 0);
  bf16x16 bphi2b = load_b(s_w2, 32, 16, 0, 16);
  // per-sample dynamic row base: 16B-aligned float4 per set element
  const f32x4* dynrow = (const f32x4*)&dyn[(s0 + row) * (NSET * DDYN)];
#pragma unroll
  for (int t = 0; t < NSET; ++t) {
    // phi1 A: lanes 0..15 own K=0..3 (one b128 load); lanes 16..31 map to K>=8 -> zero pad
    f32x4 v = {0.0f, 0.0f, 0.0f, 0.0f};
    if (lane < 16) v = dynrow[t];
    bf16x16 a;
#pragma unroll
    for (int j = 4; j < 16; ++j) a[j] = bfb(0);
    a[0] = (__bf16)v.x; a[1] = (__bf16)v.y; a[2] = (__bf16)v.z; a[3] = (__bf16)v.w;

    f32x8 c1 = wmma_bf16(a, bphi1, zc);
    store_c(myb, 32, 0, c1, s_b1, true);       // relu(phi1) -> LDS [16][16]
    bf16x16 a2 = load_a(myb, 32, 0, 16);       // transpose bounce (K pad 16->32)
    f32x8 c2 = wmma_bf16(a2, bphi2a, zc);
    relu_bias_acc(acc0, c2, s_b2, 0);
    f32x8 c3 = wmma_bf16(a2, bphi2b, zc);
    relu_bias_acc(acc1, c3, s_b2, 16);
  }
  store_c(myb, 32, 0,  acc0, nullptr, false);  // x_sum -> LDS [16][32]
  store_c(myb, 32, 16, acc1, nullptr, false);

  // ---- rho1: [16][32] x [32][32] ----
  bf16x16 ar = load_a(myb, 32, 0, 32);
  f32x8 r0 = wmma_bf16(ar, load_b(s_wr1, 32, 32, 0, 0), zc);
  f32x8 r1 = wmma_bf16(ar, load_b(s_wr1, 32, 32, 0, 16), zc);
  store_c(myb, 32, 0,  r0, s_br1, true);
  store_c(myb, 32, 16, r1, s_br1, true);

  // ---- rho2: [16][32] x [32][16] -> concat cols 0..15 ----
  bf16x16 ar2 = load_a(myb, 32, 0, 32);
  f32x8 r2 = wmma_bf16(ar2, load_b(s_wr2, 16, 32, 0, 0), zc);
  store_c(myb, 32, 0, r2, s_br2, true);
  {  // concat static_input into cols 16..22, zero 23..31
    const int c2 = 16 + (lane & 15);
    const int rb = (lane >> 4) << 3;
#pragma unroll
    for (int r = 0; r < 8; ++r) {
      unsigned short v = 0;
      if (c2 < 16 + DSTA) v = f2bf(sta[(s0 + rb + r) * DSTA + (c2 - 16)]);
      myb[(rb + r) * 32 + c2] = v;
    }
  }

  // ---- q1: [16][23(pad32)] x [23(pad32)][64] ----
  bf16x16 aq1 = load_a(myb, 32, 0, 32);
  f32x8 q1c[4];
#pragma unroll
  for (int nt = 0; nt < 4; ++nt)
    q1c[nt] = wmma_bf16(aq1, load_b(s_wq1, 64, 32, 0, nt * 16), zc);
#pragma unroll
  for (int nt = 0; nt < 4; ++nt)
    store_c(myb, 64, nt * 16, q1c[nt], s_bq1, true);

  // ---- q2: [16][64] x [64][64] (two K steps) ----
  bf16x16 aq2lo = load_a(myb, 64, 0, 64);
  bf16x16 aq2hi = load_a(myb, 64, 32, 64);
  f32x8 q2c[4];
#pragma unroll
  for (int nt = 0; nt < 4; ++nt) {
    f32x8 c = wmma_bf16(aq2lo, load_b(s_wq2, 64, 64, 0, nt * 16), zc);
    q2c[nt] = wmma_bf16(aq2hi, load_b(s_wq2, 64, 64, 32, nt * 16), c);
  }
#pragma unroll
  for (int nt = 0; nt < 4; ++nt)
    store_c(myb, 64, nt * 16, q2c[nt], s_bq2, true);

  // ---- out: [16][64] x [64][5(pad16)] ----
  bf16x16 aolo = load_a(myb, 64, 0, 64);
  bf16x16 aohi = load_a(myb, 64, 32, 64);
  f32x8 oc = wmma_bf16(aolo, load_b(s_wo, 16, 64, 0, 0), zc);
  oc = wmma_bf16(aohi, load_b(s_wo, 16, 64, 32, 0), oc);
  {
    const int col = lane & 15;
    const int rb  = (lane >> 4) << 3;
    if (col < 5) {
      float bb = s_bo[col];
#pragma unroll
      for (int r = 0; r < 8; ++r)
        __builtin_nontemporal_store(oc[r] + bb, &out[(s0 + rb + r) * 5 + col]);
    }
  }
}

extern "C" void kernel_launch(void* const* d_in, const int* in_sizes, int n_in,
                              void* d_out, int out_size, void* d_ws, size_t ws_size,
                              hipStream_t stream) {
  (void)in_sizes; (void)n_in; (void)out_size; (void)d_ws; (void)ws_size;
  const float* dyn = (const float*)d_in[0];
  const float* sta = (const float*)d_in[1];
  const float* w1  = (const float*)d_in[2];
  const float* b1  = (const float*)d_in[3];
  const float* w2  = (const float*)d_in[4];
  const float* b2  = (const float*)d_in[5];
  const float* wr1 = (const float*)d_in[6];
  const float* br1 = (const float*)d_in[7];
  const float* wr2 = (const float*)d_in[8];
  const float* br2 = (const float*)d_in[9];
  const float* wq1 = (const float*)d_in[10];
  const float* bq1 = (const float*)d_in[11];
  const float* wq2 = (const float*)d_in[12];
  const float* bq2 = (const float*)d_in[13];
  const float* wo  = (const float*)d_in[14];
  const float* bo  = (const float*)d_in[15];
  float* out = (float*)d_out;

  const int blocks = B_TOTAL / 128;   // 8 waves * 16 samples per block
  deepset_q_kernel<<<dim3(blocks), dim3(256), 0, stream>>>(
      dyn, sta, w1, b1, w2, b2, wr1, br1, wr2, br2,
      wq1, bq1, wq2, bq2, wo, bo, out);
}
